// GRATV3_27642409517709
// MI455X (gfx1250) — matrix operand order repaired and checked by reading, not compiled, not run on hardware
//
#include <hip/hip_runtime.h>
#include <hip/hip_bf16.h>
#include <math.h>

// CDNA5 / gfx1250, wave32. GAT-style 3-layer network.
// Strategy:
//   - GEMM z = act(h) @ W via v_wmma_f32_16x16x32_f16 (K=32 == feature dim,
//     so a 16x32 output row-tile = exactly 2 WMMA instructions per wave).
//     The attention dot products S[n]=z[n].a_lo, T[n]=z[n].a_hi are fused
//     into the WMMA epilogue (D tile is already in registers: per-lane FMA
//     pair + 16-lane shfl_xor tree per row).
//   - dst is sorted -> CSR row offsets by binary search, then segment
//     softmax + weighted aggregation with ONE wave per node, no atomics:
//     lane-strided shfl max / shfl sum, then lane==feature coalesced 128B
//     gathers of z[src] rows. Whole working set (~27MB) is L2-resident
//     (192MB L2), so the random edge gathers run at L2 bandwidth.

typedef __attribute__((ext_vector_type(16))) _Float16 v16h;
typedef __attribute__((ext_vector_type(8)))  float    v8f;

#define NEG_SLOPE 0.2f

__device__ __forceinline__ float act_in(float x, int mode) {
    if (mode == 1) return tanhf(x);
    if (mode == 2) return fmaxf(x, 0.0f);
    return x;
}

__device__ __forceinline__ float leaky(float x) {
    return (x > 0.0f) ? x : NEG_SLOPE * x;
}

// ---------------------------------------------------------------------------
// CSR row offsets from sorted dst: rowptr[n] = lower_bound(dst, n), rowptr[N]=E
// ---------------------------------------------------------------------------
__global__ void build_rowptr_kernel(const int* __restrict__ dst, int E,
                                    int* __restrict__ rowptr, int N) {
    int n = blockIdx.x * blockDim.x + threadIdx.x;
    if (n > N) return;
    int lo = 0, hi = E;
    while (lo < hi) {
        int mid = (lo + hi) >> 1;
        if (dst[mid] < n) lo = mid + 1; else hi = mid;
    }
    rowptr[n] = lo;
}

// ---------------------------------------------------------------------------
// Z[N,32] = act(X[N,32]) @ W[32,32], fused S[n]=Z[n].a[0:32], T[n]=Z[n].a[32:64]
// One wave per 16-row tile, 2 WMMAs (K fully consumed per instruction).
// A fragment (f16 16x32, ISA layout): lane l, m = l&15, h = l>>4.
//   element i in [0,8)  -> K = 8*h + i          (contiguous run of 8)
//   element i in [8,16) -> K = 16 + 8*h + (i-8) (contiguous run of 8)
// B fragment (f16 32x16): lane l, col n = l&15, h = l>>4; same K order as A.
// C/D (f32 16x16): lane l, col n = l&15; reg r -> row = r + 8*(l>>4).
// ---------------------------------------------------------------------------
__global__ void gemm32_wmma_kernel(const float* __restrict__ X,
                                   const float* __restrict__ W,
                                   const float* __restrict__ av,  // a [64]
                                   float* __restrict__ Z,
                                   float* __restrict__ S,
                                   float* __restrict__ T,
                                   int N, int act_mode) {
    int wave = (int)((blockIdx.x * (size_t)blockDim.x + threadIdx.x) >> 5);
    int lane = threadIdx.x & 31;
    int row0 = wave * 16;
    if (row0 >= N) return;

    int m = lane & 15;
    int hsel = lane >> 4;

    // ---- A fragment: two contiguous 8-float runs per lane (vectorizes) ----
    int r = row0 + m;
    bool rv = (r < N);
    const float* Ar = X + (size_t)(rv ? r : 0) * 32;
    v16h a;
#pragma unroll
    for (int i = 0; i < 8; ++i) {
        float x = rv ? act_in(Ar[8 * hsel + i], act_mode) : 0.0f;
        a[i] = (_Float16)x;
    }
#pragma unroll
    for (int i = 0; i < 8; ++i) {
        float x = rv ? act_in(Ar[16 + 8 * hsel + i], act_mode) : 0.0f;
        a[8 + i] = (_Float16)x;
    }

    // ---- B fragments for output columns [0,16) and [16,32) ----
    int ncol = lane & 15;
    v16h b0, b1;
#pragma unroll
    for (int i = 0; i < 16; ++i) {
        int k = (i < 8) ? (8 * hsel + i) : (16 + 8 * hsel + (i - 8));
        b0[i] = (_Float16)W[k * 32 + ncol];
        b1[i] = (_Float16)W[k * 32 + 16 + ncol];
    }

    v8f c0 = {}, c1 = {};
    c0 = __builtin_amdgcn_wmma_f32_16x16x32_f16(false, a, false, b0,
                                                (short)0, c0, false, false);
    c1 = __builtin_amdgcn_wmma_f32_16x16x32_f16(false, a, false, b1,
                                                (short)0, c1, false, false);

    // ---- store D tile: uniform bound check hoisted out of the loop ----
    if (row0 + 16 <= N) {
#pragma unroll
        for (int rr = 0; rr < 8; ++rr) {
            int row = row0 + rr + 8 * hsel;
            Z[(size_t)row * 32 + ncol]      = c0[rr];
            Z[(size_t)row * 32 + 16 + ncol] = c1[rr];
        }
    } else {
#pragma unroll
        for (int rr = 0; rr < 8; ++rr) {
            int row = row0 + rr + 8 * hsel;
            if (row < N) {
                Z[(size_t)row * 32 + ncol]      = c0[rr];
                Z[(size_t)row * 32 + 16 + ncol] = c1[rr];
            }
        }
    }

    // ---- fused attention dots: lane owns cols {n, 16+n} of 8 rows ----
    float aS0 = av[ncol],      aS1 = av[16 + ncol];
    float aT0 = av[32 + ncol], aT1 = av[48 + ncol];
#pragma unroll
    for (int rr = 0; rr < 8; ++rr) {
        float ps = c0[rr] * aS0 + c1[rr] * aS1;
        float pt = c0[rr] * aT0 + c1[rr] * aT1;
#pragma unroll
        for (int off = 8; off; off >>= 1) {   // reduce within 16-lane half
            ps += __shfl_xor(ps, off, 32);
            pt += __shfl_xor(pt, off, 32);
        }
        int row = row0 + rr + 8 * hsel;
        if (ncol == 0 && row < N) { S[row] = ps; T[row] = pt; }
    }
}

// ---------------------------------------------------------------------------
// Segment softmax + weighted aggregation, F=32. One wave per destination node.
// Pass 1/2: lane-strided max & exp-sum reductions over this node's edge range.
// Pass 3: chunk of 32 edges -> lane loads its edge's weight/src, then the wave
// iterates the chunk with broadcasts; lane==feature does coalesced 128B
// gathers of Z[src] rows.
// ---------------------------------------------------------------------------
__global__ void agg32_kernel(const float* __restrict__ Z,
                             const float* __restrict__ S,
                             const float* __restrict__ T,
                             const int* __restrict__ src,
                             const int* __restrict__ rowptr,
                             float* __restrict__ H, int N) {
    int node = (int)((blockIdx.x * (size_t)blockDim.x + threadIdx.x) >> 5);
    int lane = threadIdx.x & 31;
    if (node >= N) return;

    int beg = rowptr[node];
    int end = rowptr[node + 1];
    float tn = T[node];

    // pass 1: segment max of leaky logits
    float mx = -INFINITY;
    for (int e = beg + lane; e < end; e += 32) {
        mx = fmaxf(mx, leaky(S[src[e]] + tn));
    }
#pragma unroll
    for (int off = 16; off; off >>= 1) mx = fmaxf(mx, __shfl_xor(mx, off, 32));
    if (!isfinite(mx)) mx = 0.0f;   // empty segment: matches reference

    // pass 2: denom
    float d = 0.0f;
    for (int e = beg + lane; e < end; e += 32) {
        d += __expf(leaky(S[src[e]] + tn) - mx);
    }
#pragma unroll
    for (int off = 16; off; off >>= 1) d += __shfl_xor(d, off, 32);
    if (!(d > 0.0f)) d = 1.0f;
    float inv = 1.0f / d;

    // pass 3: weighted aggregation, lane == feature
    float acc = 0.0f;
    for (int base = beg; base < end; base += 32) {
        int cnt = end - base; if (cnt > 32) cnt = 32;
        int se = 0; float w = 0.0f;
        if (lane < cnt) {
            se = src[base + lane];
            w = __expf(leaky(S[se] + tn) - mx) * inv;
        }
        for (int j = 0; j < cnt; ++j) {
            float wj = __shfl(w, j, 32);
            int   sj = __shfl(se, j, 32);
            acc += wj * Z[(size_t)sj * 32 + lane];   // coalesced 128B row
        }
    }
    H[(size_t)node * 32 + lane] = acc;
}

// ---------------------------------------------------------------------------
// Layer 3 projection (OUT_F = 1): z3[n] = relu(H[n]) . W3, plus scalars.
// ---------------------------------------------------------------------------
__global__ void l3_dot_kernel(const float* __restrict__ H,
                              const float* __restrict__ W3,
                              const float* __restrict__ a3,
                              float* __restrict__ z3,
                              float* __restrict__ S, float* __restrict__ T,
                              int N) {
    int node = (int)((blockIdx.x * (size_t)blockDim.x + threadIdx.x) >> 5);
    int lane = threadIdx.x & 31;
    if (node >= N) return;
    float x = fmaxf(H[(size_t)node * 32 + lane], 0.0f) * W3[lane];
#pragma unroll
    for (int off = 16; off; off >>= 1) x += __shfl_xor(x, off, 32);
    if (lane == 0) {
        z3[node] = x;
        S[node]  = x * a3[0];
        T[node]  = x * a3[1];
    }
}

// ---------------------------------------------------------------------------
// Layer 3 segment softmax + scalar aggregation + sigmoid. Wave per node,
// fully lane-strided (messages are scalars).
// ---------------------------------------------------------------------------
__global__ void agg1_kernel(const float* __restrict__ z3,
                            const float* __restrict__ S,
                            const float* __restrict__ T,
                            const int* __restrict__ src,
                            const int* __restrict__ rowptr,
                            float* __restrict__ out, int N) {
    int node = (int)((blockIdx.x * (size_t)blockDim.x + threadIdx.x) >> 5);
    int lane = threadIdx.x & 31;
    if (node >= N) return;

    int beg = rowptr[node];
    int end = rowptr[node + 1];
    float tn = T[node];

    float mx = -INFINITY;
    for (int e = beg + lane; e < end; e += 32) {
        mx = fmaxf(mx, leaky(S[src[e]] + tn));
    }
#pragma unroll
    for (int off = 16; off; off >>= 1) mx = fmaxf(mx, __shfl_xor(mx, off, 32));
    if (!isfinite(mx)) mx = 0.0f;

    float d = 0.0f, num = 0.0f;
    for (int e = beg + lane; e < end; e += 32) {
        int  se = src[e];
        float w = __expf(leaky(S[se] + tn) - mx);
        d   += w;
        num += w * z3[se];
    }
#pragma unroll
    for (int off = 16; off; off >>= 1) {
        d   += __shfl_xor(d, off, 32);
        num += __shfl_xor(num, off, 32);
    }
    if (!(d > 0.0f)) d = 1.0f;
    float v = num / d;
    if (lane == 0) out[node] = 1.0f / (1.0f + __expf(-v));
}

// ---------------------------------------------------------------------------
extern "C" void kernel_launch(void* const* d_in, const int* in_sizes, int n_in,
                              void* d_out, int out_size, void* d_ws, size_t ws_size,
                              hipStream_t stream) {
    const float* feature = (const float*)d_in[0];
    const float* W1 = (const float*)d_in[1];
    const float* a1 = (const float*)d_in[2];
    const float* W2 = (const float*)d_in[3];
    const float* a2 = (const float*)d_in[4];
    const float* W3 = (const float*)d_in[5];
    const float* a3 = (const float*)d_in[6];
    const int*  src = (const int*)d_in[7];
    const int*  dst = (const int*)d_in[8];

    const int N = in_sizes[0] / 32;   // 100000
    const int E = in_sizes[7];        // 1600000

    float* out = (float*)d_out;

    // workspace carve-out (~27 MB): z | h | S | T | z3 | rowptr
    float* z      = (float*)d_ws;
    float* hbuf   = z    + (size_t)N * 32;
    float* Sbuf   = hbuf + (size_t)N * 32;
    float* Tbuf   = Sbuf + N;
    float* z3     = Tbuf + N;
    int*   rowptr = (int*)(z3 + N);

    const int BLK = 256;
    int waveGrid = (N * 32 + BLK - 1) / BLK;               // 1 wave per node
    int gemmThreads = ((N + 15) / 16) * 32;                // 1 wave per 16 rows
    int gemmGrid = (gemmThreads + BLK - 1) / BLK;

    // CSR offsets from sorted dst (topology fixed across all 3 layers)
    build_rowptr_kernel<<<(N + 1 + BLK - 1) / BLK, BLK, 0, stream>>>(dst, E, rowptr, N);

    // ----- layer 1: z = feature @ W1 (+ fused dots); h = agg(z) -----
    gemm32_wmma_kernel<<<gemmGrid, BLK, 0, stream>>>(feature, W1, a1, z,
                                                     Sbuf, Tbuf, N, /*act=*/0);
    agg32_kernel<<<waveGrid, BLK, 0, stream>>>(z, Sbuf, Tbuf, src, rowptr, hbuf, N);

    // ----- layer 2: z = tanh(h) @ W2 (+ fused dots); h = agg(z) -----
    gemm32_wmma_kernel<<<gemmGrid, BLK, 0, stream>>>(hbuf, W2, a2, z,
                                                     Sbuf, Tbuf, N, /*act=*/1);
    agg32_kernel<<<waveGrid, BLK, 0, stream>>>(z, Sbuf, Tbuf, src, rowptr, hbuf, N);

    // ----- layer 3: z3 = relu(h) @ W3 (OUT_F=1); out = sigmoid(agg(z3)) -----
    l3_dot_kernel<<<waveGrid, BLK, 0, stream>>>(hbuf, W3, a3, z3, Sbuf, Tbuf, N);
    agg1_kernel<<<waveGrid, BLK, 0, stream>>>(z3, Sbuf, Tbuf, src, rowptr, out, N);

    (void)n_in; (void)out_size; (void)ws_size;
}
